// PolynormerAttention_67894843015256
// MI455X (gfx1250) — matrix operand order repaired and checked by reading, not compiled, not run on hardware
//
#include <hip/hip_runtime.h>
#include <hip/hip_bf16.h>
#include <math.h>

typedef __bf16 bf16;
typedef __attribute__((ext_vector_type(16))) __bf16 v16bf;
typedef __attribute__((ext_vector_type(8)))  float  v8f;

#define BATCH  4
#define NTOK   16384
#define CHANS  512
#define NHEAD  8
#define HDIM   64
#define INNER  512
#define NCOLS  1536   /* Wh|Wk|Wv concatenated */
#define BETA   0.9f

union FragU { v16bf v; unsigned int u[8]; bf16 e[16]; };

// A fragment (16x32 bf16): lane l(0..15) holds row l; lanes 16..31 same rows,
// K offset by 8. VGPR0..3 = K {hi*8+0..7}, VGPR4..7 = K {16+hi*8+0..7}.
// LDS tile is row-major [row][k] so each VGPR pair is one dword load.
__device__ inline v16bf lds_load_a(const bf16* lds, int row0, int k0, int lda, int lane) {
  int l = lane & 15, hi = lane >> 4;
  const unsigned int* p = (const unsigned int*)lds;
  int base = (((row0 + l) * lda) + k0 + hi * 8) >> 1;
  FragU f;
#pragma unroll
  for (int i = 0; i < 4; ++i) f.u[i] = p[base + i];
#pragma unroll
  for (int i = 0; i < 4; ++i) f.u[4 + i] = p[base + 8 + i];
  return f.v;
}

// B fragment (32x16 bf16): lane l holds column col0+l; VGPR0..3 = K
// {k0+hi*8+0..7}, VGPR4..7 = K {k0+16+hi*8+0..7}. LDS tile is column-major [col][k] (ldk even).
__device__ inline v16bf lds_load_b(const bf16* ldsB, int col0, int k0, int ldk, int lane) {
  int l = lane & 15, hi = lane >> 4;
  const unsigned int* p = (const unsigned int*)ldsB;
  int base = (((col0 + l) * ldk) + k0 + hi * 8) >> 1;
  FragU f;
#pragma unroll
  for (int i = 0; i < 8; ++i) f.u[i] = p[base + (i & 3) + (i >> 2) * 8];
  return f.v;
}

#define WMMA_BF16(a, b, c) \
  __builtin_amdgcn_wmma_f32_16x16x32_bf16(false, (a), false, (b), (short)0, (c), false, false)

// Async global->LDS copy, 16 bytes per lane. The 24-bit inst offset is added
// to BOTH the LDS and global address, so equal-stride row copies need only
// one (lds, addr) pair. Tracked by ASYNCcnt.
#define ASYNC_B128(ldsOff, gAddr, OFF)                                        \
  asm volatile("global_load_async_to_lds_b128 %0, %1, off offset:" OFF        \
               :: "v"(ldsOff), "v"(gAddr) : "memory")
#define ASYNC_WAIT() asm volatile("s_wait_asynccnt 0x0" ::: "memory")

__device__ inline unsigned lds_off32(const void* p) {
  return (unsigned)(unsigned long long)p;   // LDS aperture: addr[31:0] = offset
}

// ---------------------------------------------------------------- weights cast
__global__ void __launch_bounds__(256)
k_wconv(const float* __restrict__ Wh, const float* __restrict__ Wk,
        const float* __restrict__ Wv, const float* __restrict__ Wo,
        bf16* __restrict__ wcat, bf16* __restrict__ wob) {
  int idx = blockIdx.x * 256 + threadIdx.x;
  if (idx < CHANS * NCOLS) {
    int r = idx / NCOLS, c = idx % NCOLS;
    float v = (c < 512) ? Wh[r * 512 + c]
            : (c < 1024) ? Wk[r * 512 + (c - 512)]
                         : Wv[r * 512 + (c - 1024)];
    wcat[idx] = (bf16)v;
  } else {
    int j = idx - CHANS * NCOLS;
    if (j < INNER * INNER) wob[j] = (bf16)Wo[j];
  }
}

__global__ void __launch_bounds__(256)
k_zero(float* __restrict__ p, int n) {
  int i = blockIdx.x * 256 + threadIdx.x;
  if (i < n) p[i] = 0.f;
}

// -------------------------------------------------- fused x@[Wh|Wk|Wv] GEMM
// 128x64 tile, 128 threads (4 waves), K-step 32. Epilogue routes columns.
__global__ void __launch_bounds__(128)
k_gemm_qkvh(const float* __restrict__ x, const bf16* __restrict__ wcat,
            const float* __restrict__ bh, const unsigned char* __restrict__ mask,
            bf16* __restrict__ hbuf, bf16* __restrict__ kbuf, bf16* __restrict__ vbuf) {
  __shared__ bf16 sA[128 * 32];
  __shared__ bf16 sB[64 * 32];
  int tid = threadIdx.x, lane = tid & 31, wv = tid >> 5;
  int row0 = blockIdx.x * 128;   // flat token row (b*N+n)
  int col0 = blockIdx.y * 64;    // 0..1535
  v8f acc[2][4] = {};

  for (int kt = 0; kt < CHANS; kt += 32) {
    { // stage A: thread t owns row t, 32 f32 -> bf16
      const float4* src = (const float4*)(x + (size_t)(row0 + tid) * CHANS + kt);
      bf16* dst = sA + tid * 32;
#pragma unroll
      for (int i = 0; i < 8; ++i) {
        float4 f = src[i];
        dst[i * 4 + 0] = (bf16)f.x; dst[i * 4 + 1] = (bf16)f.y;
        dst[i * 4 + 2] = (bf16)f.z; dst[i * 4 + 3] = (bf16)f.w;
      }
      if (kt + 32 < CHANS)
        __builtin_prefetch(x + (size_t)(row0 + tid) * CHANS + kt + 32, 0, 3);
    }
    { // stage B col-major: thread t -> weight row kt+r, 16 cols
      int r = tid >> 2, cseg = (tid & 3) * 16;
      const bf16* src = wcat + (size_t)(kt + r) * NCOLS + col0 + cseg;
#pragma unroll
      for (int j = 0; j < 16; ++j) sB[(cseg + j) * 32 + r] = src[j];
    }
    __syncthreads();
    v16bf a0 = lds_load_a(sA, wv * 32,      0, 32, lane);
    v16bf a1 = lds_load_a(sA, wv * 32 + 16, 0, 32, lane);
#pragma unroll
    for (int j = 0; j < 4; ++j) {
      v16bf b = lds_load_b(sB, j * 16, 0, 32, lane);
      acc[0][j] = WMMA_BF16(a0, b, acc[0][j]);
      acc[1][j] = WMMA_BF16(a1, b, acc[1][j]);
    }
    __syncthreads();
  }

  int l = lane & 15, hi = lane >> 4;
#pragma unroll
  for (int i = 0; i < 2; ++i)
#pragma unroll
    for (int j = 0; j < 4; ++j)
#pragma unroll
      for (int r = 0; r < 8; ++r) {
        int gr = row0 + wv * 32 + i * 16 + hi * 8 + r;
        int gc = col0 + j * 16 + l;
        float a = acc[i][j][r];
        if (gc < 512) {                       // h = x@Wh + bh
          hbuf[(size_t)gr * INNER + gc] = (bf16)(a + bh[gc]);
        } else if (gc < 1024) {               // k = sigmoid(x@Wk), per-head layout
          int c = gc - 512, d = c >> 3, hh = c & 7;
          int b = gr >> 14, n = gr & (NTOK - 1);
          float s = 1.f / (1.f + __expf(-a));
          kbuf[((size_t)(b * NHEAD + hh) * NTOK + n) * HDIM + d] = (bf16)s;
        } else {                              // v = mask ? x@Wv : 0
          int c = gc - 1024, d = c >> 3, hh = c & 7;
          int b = gr >> 14, n = gr & (NTOK - 1);
          float mv = mask[gr] ? a : 0.f;
          vbuf[((size_t)(b * NHEAD + hh) * NTOK + n) * HDIM + d] = (bf16)mv;
        }
      }
}

// ---------------------------------------------------------------- ksum = sum_n k
__global__ void __launch_bounds__(256)
k_ksum(const bf16* __restrict__ kbuf, float* __restrict__ ksum) {
  __shared__ float s[256];
  int bh = blockIdx.x, t = threadIdx.x;
  int d = t & 63, part = t >> 6;
  const bf16* kp = kbuf + (size_t)bh * NTOK * HDIM;
  float acc = 0.f;
  for (int n = part * 4096; n < (part + 1) * 4096; ++n)
    acc += (float)kp[(size_t)n * HDIM + d];
  s[t] = acc;
  __syncthreads();
  if (t < 64) ksum[bh * HDIM + t] = s[t] + s[t + 64] + s[t + 128] + s[t + 192];
}

// ---------------------------------------------- kv[d,m] = sum_n k[n,d] v[n,m]
// Both operands staged transposed in LDS ([d][n] / [m][n]) so A and B fragment
// loads are contiguous-pair ds_load_b128s; 64-deep n slabs = 8 WMMA per barrier.
#define K2_CHUNK 2048
__global__ void __launch_bounds__(128)
k_gemm_kv(const bf16* __restrict__ kbuf, const bf16* __restrict__ vbuf,
          float* __restrict__ kv) {
  __shared__ bf16 sKT[64 * 64];  // [d][n]
  __shared__ bf16 sV[64 * 64];   // [m][n]
  int tid = threadIdx.x, lane = tid & 31, wv = tid >> 5;
  int bh = blockIdx.y;
  int n0 = blockIdx.x * K2_CHUNK;
  const bf16* kp = kbuf + (size_t)bh * NTOK * HDIM;
  const bf16* vp = vbuf + (size_t)bh * NTOK * HDIM;
  v8f acc[4] = {};

  for (int nt = 0; nt < K2_CHUNK; nt += 64) {
    int r = tid >> 1;             // n row 0..63
    int dseg = (tid & 1) * 32;    // half of d/m range
    const bf16* ks = kp + (size_t)(n0 + nt + r) * HDIM + dseg;
    const bf16* vs = vp + (size_t)(n0 + nt + r) * HDIM + dseg;
#pragma unroll
    for (int j = 0; j < 32; ++j) {
      sKT[(dseg + j) * 64 + r] = ks[j];
      sV[(dseg + j) * 64 + r]  = vs[j];
    }
    __syncthreads();
#pragma unroll
    for (int kt = 0; kt < 64; kt += 32) {
      v16bf b = lds_load_b(sV, wv * 16, kt, 64, lane);
#pragma unroll
      for (int i = 0; i < 4; ++i) {
        v16bf a = lds_load_a(sKT, i * 16, kt, 64, lane);  // A = k^T
        acc[i] = WMMA_BF16(a, b, acc[i]);
      }
    }
    __syncthreads();
  }

  int l = lane & 15, hi = lane >> 4;
#pragma unroll
  for (int i = 0; i < 4; ++i)
#pragma unroll
    for (int r = 0; r < 8; ++r) {
      int d = i * 16 + hi * 8 + r;
      int m = wv * 16 + l;
      atomicAdd(&kv[(size_t)bh * 4096 + d * 64 + m], acc[i][r]);
    }
}

// ---------------------------------------------------------------- den = q.ksum
__global__ void __launch_bounds__(256)
k_den(const bf16* __restrict__ kbuf, const float* __restrict__ ksum,
      float* __restrict__ den) {
  int g = blockIdx.x * 256 + threadIdx.x;   // (b*N+n)*8 + h
  int nf = g >> 3, h = g & 7;
  int b = nf >> 14, n = nf & (NTOK - 1);
  const bf16*  q  = kbuf + ((size_t)(b * NHEAD + h) * NTOK + n) * HDIM;
  const float* ks = ksum + (b * NHEAD + h) * HDIM;
  float acc = 0.f;
#pragma unroll
  for (int d = 0; d < HDIM; ++d) acc += (float)q[d] * ks[d];
  den[g] = acc;
}

// -------------------------------------------- num = q @ kv, scaled by 1/den
__global__ void __launch_bounds__(128)
k_gemm_num(const bf16* __restrict__ kbuf, const float* __restrict__ kv,
           const float* __restrict__ den, bf16* __restrict__ ubuf) {
  __shared__ bf16 sA[128 * 64];   // q tile [n][d]
  __shared__ bf16 sB[64 * 64];    // kv col-major [m][d]
  int tid = threadIdx.x, lane = tid & 31, wv = tid >> 5;
  int bh = blockIdx.y;
  int row0 = blockIdx.x * 128;
  const bf16* kp = kbuf + (size_t)bh * NTOK * HDIM;

  { // stage A (full K=64) asynchronously: row t = 128B, equal strides
    unsigned la = lds_off32(sA + (size_t)tid * 64);
    unsigned long long ga =
        (unsigned long long)(kp + (size_t)(row0 + tid) * HDIM);
    ASYNC_B128(la, ga, "0");   ASYNC_B128(la, ga, "16");
    ASYNC_B128(la, ga, "32");  ASYNC_B128(la, ga, "48");
    ASYNC_B128(la, ga, "64");  ASYNC_B128(la, ga, "80");
    ASYNC_B128(la, ga, "96");  ASYNC_B128(la, ga, "112");
  }
  { // stage B: kv f32 [d][m] -> bf16 col-major [m][d]
#pragma unroll
    for (int i = 0; i < 32; ++i) {
      int idx = tid * 32 + i;
      int d = idx >> 6, m = idx & 63;
      sB[m * 64 + d] = (bf16)kv[(size_t)bh * 4096 + idx];
    }
  }
  ASYNC_WAIT();
  __syncthreads();

  v8f acc[2][4] = {};
#pragma unroll
  for (int kt = 0; kt < 64; kt += 32) {
    v16bf a0 = lds_load_a(sA, wv * 32,      kt, 64, lane);
    v16bf a1 = lds_load_a(sA, wv * 32 + 16, kt, 64, lane);
#pragma unroll
    for (int j = 0; j < 4; ++j) {
      v16bf b = lds_load_b(sB, j * 16, kt, 64, lane);
      acc[0][j] = WMMA_BF16(a0, b, acc[0][j]);
      acc[1][j] = WMMA_BF16(a1, b, acc[1][j]);
    }
  }

  int l = lane & 15, hi = lane >> 4;
  int b = bh >> 3, h = bh & 7;
#pragma unroll
  for (int i = 0; i < 2; ++i)
#pragma unroll
    for (int j = 0; j < 4; ++j)
#pragma unroll
      for (int r = 0; r < 8; ++r) {
        int n = row0 + wv * 32 + i * 16 + hi * 8 + r;
        int m = j * 16 + l;
        float dv = den[(size_t)(b * NTOK + n) * NHEAD + h] + 1e-6f;
        float val = acc[i][j][r] / dv;
        ubuf[(size_t)(b * NTOK + n) * INNER + m * NHEAD + h] = (bf16)val;
      }
}

// ------------------------------------- layernorm(u) * (h + beta), in place
__global__ void __launch_bounds__(256)
k_ln(bf16* __restrict__ ubuf, const bf16* __restrict__ hbuf,
     const float* __restrict__ g, const float* __restrict__ bb) {
  __shared__ float ssum[256], ssq[256];
  int row = blockIdx.x, t = threadIdx.x;
  bf16* up = ubuf + (size_t)row * INNER;
  float x0 = (float)up[t], x1 = (float)up[t + 256];
  ssum[t] = x0 + x1;
  ssq[t]  = x0 * x0 + x1 * x1;
  __syncthreads();
  for (int s = 128; s > 0; s >>= 1) {
    if (t < s) { ssum[t] += ssum[t + s]; ssq[t] += ssq[t + s]; }
    __syncthreads();
  }
  float mu  = ssum[0] * (1.f / 512.f);
  float var = ssq[0] * (1.f / 512.f) - mu * mu;
  float rs  = rsqrtf(var + 1e-5f);
  const bf16* hp = hbuf + (size_t)row * INNER;
  float y0 = ((x0 - mu) * rs * g[t]       + bb[t])       * ((float)hp[t]       + BETA);
  float y1 = ((x1 - mu) * rs * g[t + 256] + bb[t + 256]) * ((float)hp[t + 256] + BETA);
  up[t]       = (bf16)y0;
  up[t + 256] = (bf16)y1;
}

// ---------------------------------------------- out = relu(u @ Wo + bo), f32
__global__ void __launch_bounds__(128)
k_gemm_out(const bf16* __restrict__ u, const bf16* __restrict__ wob,
           const float* __restrict__ bo, float* __restrict__ out) {
  __shared__ bf16 sA[128 * 32];
  __shared__ bf16 sB[64 * 32];
  int tid = threadIdx.x, lane = tid & 31, wv = tid >> 5;
  int row0 = blockIdx.x * 128, col0 = blockIdx.y * 64;
  v8f acc[2][4] = {};

  for (int kt = 0; kt < INNER; kt += 32) {
    { // stage A (already bf16) asynchronously: row t = 64B, equal strides
      unsigned la = lds_off32(sA + (size_t)tid * 32);
      unsigned long long ga =
          (unsigned long long)(u + (size_t)(row0 + tid) * INNER + kt);
      ASYNC_B128(la, ga, "0");
      ASYNC_B128(la, ga, "16");
    }
    {
      int r = tid >> 2, cseg = (tid & 3) * 16;
      const bf16* src = wob + (size_t)(kt + r) * INNER + col0 + cseg;
#pragma unroll
      for (int j = 0; j < 16; ++j) sB[(cseg + j) * 32 + r] = src[j];
    }
    ASYNC_WAIT();
    __syncthreads();
    v16bf a0 = lds_load_a(sA, wv * 32,      0, 32, lane);
    v16bf a1 = lds_load_a(sA, wv * 32 + 16, 0, 32, lane);
#pragma unroll
    for (int j = 0; j < 4; ++j) {
      v16bf b = lds_load_b(sB, j * 16, 0, 32, lane);
      acc[0][j] = WMMA_BF16(a0, b, acc[0][j]);
      acc[1][j] = WMMA_BF16(a1, b, acc[1][j]);
    }
    __syncthreads();
  }

  int l = lane & 15, hi = lane >> 4;
#pragma unroll
  for (int i = 0; i < 2; ++i)
#pragma unroll
    for (int j = 0; j < 4; ++j)
#pragma unroll
      for (int r = 0; r < 8; ++r) {
        int gr = row0 + wv * 32 + i * 16 + hi * 8 + r;
        int gc = col0 + j * 16 + l;
        out[(size_t)gr * INNER + gc] = fmaxf(acc[i][j][r] + bo[gc], 0.f);
      }
}

// ------------------------------------------------------------------ launcher
extern "C" void kernel_launch(void* const* d_in, const int* in_sizes, int n_in,
                              void* d_out, int out_size, void* d_ws, size_t ws_size,
                              hipStream_t stream) {
  const float* x          = (const float*)d_in[0];
  const unsigned char* mk = (const unsigned char*)d_in[1];
  const float* Wh  = (const float*)d_in[2];
  const float* bh  = (const float*)d_in[3];
  const float* Wk  = (const float*)d_in[4];
  const float* Wv  = (const float*)d_in[5];
  const float* lng = (const float*)d_in[6];
  const float* lnb = (const float*)d_in[7];
  const float* Wo  = (const float*)d_in[8];
  const float* bo  = (const float*)d_in[9];
  float* out = (float*)d_out;
  char* ws = (char*)d_ws;

  const size_t M = (size_t)BATCH * NTOK;           // 65536 token rows
  size_t off = 0;
  bf16*  wcat = (bf16*)(ws + off);  off += (size_t)CHANS * NCOLS * 2;   // 1.5 MB
  bf16*  wob  = (bf16*)(ws + off);  off += (size_t)INNER * INNER * 2;   // 0.5 MB
  bf16*  hbuf = (bf16*)(ws + off);  off += M * INNER * 2;               // 64 MB
  bf16*  kbuf = (bf16*)(ws + off);  off += M * INNER * 2;               // 64 MB  [B,H,N,64]
  bf16*  vbuf = (bf16*)(ws + off);  off += M * INNER * 2;               // 64 MB  [B,H,N,64]
  float* kv   = (float*)(ws + off); off += (size_t)32 * 64 * 64 * 4;    // 512 KB [B*H,64,64]
  float* ksum = (float*)(ws + off); off += (size_t)32 * 64 * 4;         // 8 KB
  float* den  = (float*)(ws + off); off += M * NHEAD * 4;               // 2 MB
  bf16*  ubuf = (bf16*)(ws + off);  off += M * INNER * 2;               // 64 MB

  // 1) cast weights to bf16 (concat Wh|Wk|Wv, plus Wo)
  k_wconv<<<(CHANS * NCOLS + INNER * INNER) / 256, 256, 0, stream>>>(
      Wh, Wk, Wv, Wo, wcat, wob);

  // 2) zero kv + ksum accumulators (contiguous in ws)
  {
    int nz = 32 * 64 * 64 + 32 * 64;
    k_zero<<<(nz + 255) / 256, 256, 0, stream>>>(kv, nz);
  }

  // 3) fused h / k / v GEMM
  {
    dim3 grid(M / 128, NCOLS / 64);
    k_gemm_qkvh<<<grid, 128, 0, stream>>>(x, wcat, bh, mk, hbuf, kbuf, vbuf);
  }

  // 4) ksum
  k_ksum<<<BATCH * NHEAD, 256, 0, stream>>>(kbuf, ksum);

  // 5) kv = k^T v   (atomic f32 accumulation over N chunks)
  {
    dim3 grid(NTOK / K2_CHUNK, BATCH * NHEAD);
    k_gemm_kv<<<grid, 128, 0, stream>>>(kbuf, vbuf, kv);
  }

  // 6) den = q . ksum
  k_den<<<(int)(M * NHEAD / 256), 256, 0, stream>>>(kbuf, ksum, den);

  // 7) num = q @ kv, scaled by 1/(den+eps) -> ubuf
  {
    dim3 grid(NTOK / 128, BATCH * NHEAD);
    k_gemm_num<<<grid, 128, 0, stream>>>(kbuf, kv, den, ubuf);
  }

  // 8) layernorm * (h + beta), in place on ubuf
  k_ln<<<(int)M, 256, 0, stream>>>(ubuf, hbuf, lng, lnb);

  // 9) out = relu(u @ Wo + bo)
  {
    dim3 grid(M / 128, INNER / 64);
    k_gemm_out<<<grid, 128, 0, stream>>>(ubuf, wob, bo, out);
  }
}
